// AttentionRNN_55508157334111
// MI455X (gfx1250) — matrix operand order repaired and checked by reading, not compiled, not run on hardware
//
#include <hip/hip_runtime.h>
#include <hip/hip_bf16.h>
#include <math.h>

#define VOCAB 50257
#define HDIM  512
#define TLEN  1024

typedef _Float16 f16;
typedef __attribute__((ext_vector_type(16))) _Float16 v16h;
typedef __attribute__((ext_vector_type(8)))  float    v8f;

#define WMMA_F16(a, b, c) \
  __builtin_amdgcn_wmma_f32_16x16x32_f16(false, (a), false, (b), (short)0, (c), false, false)

// ---------------------------------------------------------------------------
// Fragment loaders (wave32, v_wmma_f32_16x16x32_f16 layouts per CDNA5 ISA §7.12.2)
// A (16x32, MxK): lane m(0-15) row m holds K = 0..7,16..23 ; lanes 16-31 same
//                 rows, K = 8..15,24..31.  -> two contiguous 16B chunks.
// B (32x16, KxN): lane l(0-15) column l holds K = 0..7,16..23 ; lanes 16-31 column
//                 l-16 hold K = 8..15,24..31.    -> two contiguous 16B chunks.
// ---------------------------------------------------------------------------
__device__ __forceinline__ v16h frag_a_rowmajor(const f16* base, int ld, int row, int k, int lane) {
  const f16* p = base + (size_t)row * ld + k + 8 * (lane >> 4);
  union { v16h v; float4 q[2]; } u;
  u.q[0] = *(const float4*)(p);
  u.q[1] = *(const float4*)(p + 16);
  return u.v;
}
// B fragment where each WMMA-column n is K-contiguous at (base + n*ld):
// works both for A@A^T (rows of H are columns of H^T) and for LDS-transposed tiles.
__device__ __forceinline__ v16h frag_b_kcontig(const f16* base, int ld, int col0, int k, int lane) {
  const f16* p = base + (size_t)(col0 + (lane & 15)) * ld + k + 8 * (lane >> 4);
  union { v16h v; float4 q[2]; } u;
  u.q[0] = *(const float4*)(p);
  u.q[1] = *(const float4*)(p + 16);
  return u.v;
}

// ---------------------------------------------------------------------------
// Kernel A: sequential RNN scan, single workgroup, h lives in LDS.
// 256 threads x 2 columns each: every h[k] broadcast and every coalesced
// float2 W-load feeds two FMAs (packed f32 pairs -> v_pk_fma_f32 / VOPD).
// states[t][0:512] = (f16) h_t
// ---------------------------------------------------------------------------
__global__ void __launch_bounds__(256)
rnn_scan_kernel(const int* __restrict__ tokens, const float* __restrict__ h0,
                const float* __restrict__ emb_tab, const float* __restrict__ Whh,
                const float* __restrict__ bh, f16* __restrict__ states) {
  __shared__ __align__(16) float h[HDIM];
  const int j2 = threadIdx.x * 2;           // columns j2, j2+1
  *(float2*)(h + j2) = *(const float2*)(h0 + j2);
  const float2 bj = *(const float2*)(bh + j2);
  __syncthreads();

  for (int t = 0; t < TLEN; ++t) {
    const int tok = tokens[t];
    float2 acc = *(const float2*)(emb_tab + (size_t)tok * HDIM + j2);
    acc.x += bj.x;
    acc.y += bj.y;
    const float4* h4 = (const float4*)h;
    const float* Wp = Whh + j2;             // W[k][j2], row stride HDIM
#pragma unroll 8
    for (int k4 = 0; k4 < HDIM / 4; ++k4) {
      const float4 hv = h4[k4];
      const float2 w0 = *(const float2*)(Wp + (size_t)(4 * k4 + 0) * HDIM);
      const float2 w1 = *(const float2*)(Wp + (size_t)(4 * k4 + 1) * HDIM);
      const float2 w2 = *(const float2*)(Wp + (size_t)(4 * k4 + 2) * HDIM);
      const float2 w3 = *(const float2*)(Wp + (size_t)(4 * k4 + 3) * HDIM);
      acc.x = fmaf(hv.x, w0.x, acc.x);  acc.y = fmaf(hv.x, w0.y, acc.y);
      acc.x = fmaf(hv.y, w1.x, acc.x);  acc.y = fmaf(hv.y, w1.y, acc.y);
      acc.x = fmaf(hv.z, w2.x, acc.x);  acc.y = fmaf(hv.z, w2.y, acc.y);
      acc.x = fmaf(hv.w, w3.x, acc.x);  acc.y = fmaf(hv.w, w3.y, acc.y);
    }
    float2 hn;
    hn.x = tanhf(acc.x);
    hn.y = tanhf(acc.y);
    __syncthreads();                        // all reads of old h complete
    *(float2*)(h + j2) = hn;
    union { f16 a[2]; unsigned u; } pk;     // packed f16x2 store (b32, aligned)
    pk.a[0] = (f16)hn.x;
    pk.a[1] = (f16)hn.y;
    *(unsigned*)(states + (size_t)t * (2 * HDIM) + j2) = pk.u;
    __syncthreads();                        // new h visible
  }
}

// ---------------------------------------------------------------------------
// Kernel B1: S = H @ H^T  (f16 WMMA, f32 out).  Upper tiles (s_tile > t_tile)
// skipped entirely — softmax never reads them.
// ---------------------------------------------------------------------------
__global__ void __launch_bounds__(128)
scores_kernel(const f16* __restrict__ states, float* __restrict__ S) {
  const int lane = threadIdx.x & 31;
  const int wave = threadIdx.x >> 5;
  const int t_tile = blockIdx.x;                 // 0..63
  const int s_tile = blockIdx.y * 4 + wave;      // 0..63
  if (s_tile > t_tile) return;                   // wave-uniform branch
  const int t0 = t_tile * 16, s0 = s_tile * 16;

  v8f c = {};
#pragma unroll
  for (int k = 0; k < HDIM; k += 32) {
    v16h a = frag_a_rowmajor(states, 2 * HDIM, t0 + (lane & 15), k, lane);
    v16h b = frag_b_kcontig (states, 2 * HDIM, s0, k, lane);   // rows of H == cols of H^T
    c = WMMA_F16(a, b, c);
  }
  const int m = t0 + 8 * (lane >> 4);
  const int n = s0 + (lane & 15);
#pragma unroll
  for (int r = 0; r < 8; ++r) S[(size_t)(m + r) * TLEN + n] = c[r];
}

// ---------------------------------------------------------------------------
// Kernel B2: causal row softmax.  P[t][s] = softmax_{s<t}(S[t][s]), else 0.
// Row t = 0 -> all zeros (matches reference ctx_0 = 0).
// ---------------------------------------------------------------------------
__global__ void __launch_bounds__(256)
softmax_kernel(const float* __restrict__ S, f16* __restrict__ P) {
  const int t = blockIdx.x;
  const int tid = threadIdx.x;
  __shared__ float red[8];
  const float* row = S + (size_t)t * TLEN;

  float m = -3.0e38f;
  for (int s = tid; s < t; s += 256) m = fmaxf(m, row[s]);
#pragma unroll
  for (int off = 16; off; off >>= 1) m = fmaxf(m, __shfl_xor(m, off, 32));
  if ((tid & 31) == 0) red[tid >> 5] = m;
  __syncthreads();
  float mm = red[0];
#pragma unroll
  for (int i = 1; i < 8; ++i) mm = fmaxf(mm, red[i]);
  __syncthreads();

  float sum = 0.f;
  for (int s = tid; s < t; s += 256) sum += __expf(row[s] - mm);
#pragma unroll
  for (int off = 16; off; off >>= 1) sum += __shfl_xor(sum, off, 32);
  if ((tid & 31) == 0) red[tid >> 5] = sum;
  __syncthreads();
  float ss = 0.f;
#pragma unroll
  for (int i = 0; i < 8; ++i) ss += red[i];
  const float inv = (t > 0) ? 1.0f / ss : 0.0f;

  f16* prow = P + (size_t)t * TLEN;
  for (int s = tid; s < TLEN; s += 256)
    prow[s] = (f16)((s < t) ? __expf(row[s] - mm) * inv : 0.0f);
}

// ---------------------------------------------------------------------------
// Kernel B3: ctx = P @ H  -> states[t][512:1024].  B (rows of H along K)
// staged transposed in LDS: Bt[n][k], row stride 40 halves (80B) for
// conflict-free ds_load_b128 fragment reads.
// ---------------------------------------------------------------------------
#define BT_LD 40
__global__ void __launch_bounds__(256)
ctx_kernel(const f16* __restrict__ P, f16* __restrict__ states) {
  __shared__ __align__(16) f16 Bt[64 * BT_LD];
  const int tid = threadIdx.x, lane = tid & 31, wave = tid >> 5;
  const int n0 = blockIdx.x * 64;        // hidden-dim tile (0..448)
  const int mw = blockIdx.y * 128 + wave * 16;

  v8f c[4] = {};
  const int nn = tid & 63, kk0 = tid >> 6;

  for (int k0 = 0; k0 < TLEN; k0 += 32) {
    __syncthreads();
#pragma unroll
    for (int it = 0; it < 8; ++it) {      // stage 32(K) x 64(N) f16, coalesced
      const int kk = kk0 + it * 4;
      Bt[nn * BT_LD + kk] = states[(size_t)(k0 + kk) * (2 * HDIM) + n0 + nn];
    }
    __syncthreads();

    v16h a = frag_a_rowmajor(P, TLEN, mw + (lane & 15), k0, lane);
#pragma unroll
    for (int ns = 0; ns < 4; ++ns) {
      v16h b = frag_b_kcontig(Bt, BT_LD, ns * 16, 0, lane);
      c[ns] = WMMA_F16(a, b, c[ns]);
    }
  }

  const int mrow = mw + 8 * (lane >> 4);
#pragma unroll
  for (int ns = 0; ns < 4; ++ns) {
    const int n = n0 + ns * 16 + (lane & 15);
#pragma unroll
    for (int r = 0; r < 8; ++r)
      states[(size_t)(mrow + r) * (2 * HDIM) + HDIM + n] = (f16)c[ns][r];
  }
}

// ---------------------------------------------------------------------------
// Kernel C: out = states @ W + bias.  M=1024, N=50257, K=1024.
// 512 threads / 256 M-rows per block (halves re-reads of the 206MB weight),
// register double-buffered B tile: next K-slice's global loads issue before
// the WMMA section so their latency hides behind the matrix ops.
// ---------------------------------------------------------------------------
__global__ void __launch_bounds__(512)
output_gemm_kernel(const f16* __restrict__ states, const float* __restrict__ W,
                   const float* __restrict__ bias, float* __restrict__ out) {
  __shared__ __align__(16) f16 Bt[64 * BT_LD];
  const int tid = threadIdx.x, lane = tid & 31, wave = tid >> 5;
  const int n0 = blockIdx.x * 64;
  const int mw = blockIdx.y * 256 + wave * 16;

  v8f c[4] = {};
  const int nn = tid & 63, kk0 = tid >> 6;       // kk0 in 0..7
  const bool nvalid = (n0 + nn) < VOCAB;
  const float* Wcol = W + n0 + nn;

  float v[4];                                    // slice k = kk0 + it*8
#pragma unroll
  for (int it = 0; it < 4; ++it)
    v[it] = nvalid ? Wcol[(size_t)(kk0 + it * 8) * VOCAB] : 0.0f;

  for (int k0 = 0; k0 < 2 * HDIM; k0 += 32) {
    __syncthreads();                             // Bt free (prev WMMAs done)
#pragma unroll
    for (int it = 0; it < 4; ++it)
      Bt[nn * BT_LD + kk0 + it * 8] = (f16)v[it];
    if (k0 + 32 < 2 * HDIM) {                    // issue next tile's loads now;
#pragma unroll                                   // waited on only next iteration
      for (int it = 0; it < 4; ++it)
        v[it] = nvalid ? Wcol[(size_t)(k0 + 32 + kk0 + it * 8) * VOCAB] : 0.0f;
    }
    __syncthreads();                             // Bt filled

    v16h a = frag_a_rowmajor(states, 2 * HDIM, mw + (lane & 15), k0, lane);
#pragma unroll
    for (int ns = 0; ns < 4; ++ns) {
      v16h b = frag_b_kcontig(Bt, BT_LD, ns * 16, 0, lane);
      c[ns] = WMMA_F16(a, b, c[ns]);
    }
  }

  const int mrow = mw + 8 * (lane >> 4);
#pragma unroll
  for (int ns = 0; ns < 4; ++ns) {
    const int n = n0 + ns * 16 + (lane & 15);
    if (n < VOCAB) {
      const float bv = bias[n];
#pragma unroll
      for (int r = 0; r < 8; ++r)
        out[(size_t)(mrow + r) * VOCAB + n] = c[ns][r] + bv;
    }
  }
}

// ---------------------------------------------------------------------------
extern "C" void kernel_launch(void* const* d_in, const int* in_sizes, int n_in,
                              void* d_out, int out_size, void* d_ws, size_t ws_size,
                              hipStream_t stream) {
  const int*   tokens = (const int*)  d_in[0];
  const float* h0     = (const float*)d_in[1];
  const float* embtab = (const float*)d_in[2];
  const float* Whh    = (const float*)d_in[3];
  const float* bh     = (const float*)d_in[4];
  const float* Wout   = (const float*)d_in[5];
  const float* bout   = (const float*)d_in[6];
  float* out = (float*)d_out;

  // ws layout: states f16 [1024][1024] (2MB) | S f32 [1024][1024] (4MB) | P f16 (2MB)
  char* ws = (char*)d_ws;
  f16*   states = (f16*)ws;
  float* S      = (float*)(ws + (2u << 20));
  f16*   P      = (f16*)(ws + (6u << 20));

  rnn_scan_kernel<<<1, 256, 0, stream>>>(tokens, h0, embtab, Whh, bh, states);
  scores_kernel<<<dim3(TLEN / 16, 16), 128, 0, stream>>>(states, S);
  softmax_kernel<<<TLEN, 256, 0, stream>>>(S, P);
  ctx_kernel<<<dim3(HDIM / 64, TLEN / 128), 256, 0, stream>>>(P, states);
  output_gemm_kernel<<<dim3((VOCAB + 63) / 64, TLEN / 256), 512, 0, stream>>>(
      states, Wout, bout, out);
}
